// RNN_67138928771392
// MI455X (gfx1250) — compile-verified, matrix operand order checked by reading, not run on hardware
//
#include <hip/hip_runtime.h>
#include <hip/hip_bf16.h>

// ---------------------------------------------------------------------------
// CDNA5 (gfx1250) bidirectional tanh-RNN forward.
// GEMMs: v_wmma_f32_16x16x32_f16 (f16 in, f32 accum), fragment-major packed
// operands so inner loops are contiguous b128 loads. The input-projection
// GEMM stages its block-shared A strip into LDS with the Tensor Data Mover
// (tensor_load_to_lds + s_wait_tensorcnt), since all 8 waves of a block share
// one mT strip (8x global-traffic reduction on the FLOP-dominant kernels).
// ---------------------------------------------------------------------------

typedef __attribute__((ext_vector_type(16))) _Float16 v16h;
typedef __attribute__((ext_vector_type(8)))  _Float16 v8h;
typedef __attribute__((ext_vector_type(8)))  float    v8f;
typedef __attribute__((ext_vector_type(4)))  unsigned int u32x4;
typedef __attribute__((ext_vector_type(4)))  int      i32x4;
typedef __attribute__((ext_vector_type(8)))  int      i32x8;

#define T_LEN 512
#define BATCH 64
#define EMB_D 300
#define EMB_P 320            // E padded to K multiple of 32
#define HID   512
#define M_TOT (T_LEN*BATCH)  // 32768 rows
#define MT    (M_TOT/16)     // 2048 M-tiles
#define NT_H  (HID/16)       // 32 N-tiles

// ISA 7.12.2: 16-bit A/B fragment: lane = (row&15) + 16*hi, element e (0..15)
// maps to K = 2v + (v>=4?8:0) + 8*hi + lo,  v=e>>1, lo=e&1.
__device__ __forceinline__ int frag_k(int e, int hi) {
    int v = e >> 1, lo = e & 1;
    return 2*v + ((v >= 4) ? 8 : 0) + 8*hi + lo;
}
// inverse: k in [0,32) -> (hi, e)
__device__ __forceinline__ void frag_k_inv(int kin, int& hi, int& e) {
    int g = kin >> 3;                  // 0:{0-7} 1:{8-15} 2:{16-23} 3:{24-31}
    hi = g & 1;
    int vbase = (g >= 2) ? 4 : 0;
    int v = vbase + ((kin & 7) >> 1);
    e = 2*v + (kin & 1);
}

// ---------------------------------------------------------------------------
// Tensor Data Mover: DMA a contiguous strip of nHalf f16 elements from global
// memory into LDS. Degenerate-2D D# per ISA ch.8: group0 = {count=1, lds_addr,
// global_addr[56:0], type=2}; group1 = {data_size=2B, tensor_dim0=nHalf,
// tensor_dim1=1, tile_dim0=nHalf (<=65535), tile_dim1/2=0(unused),
// tensor_dim0_stride=nHalf}. Caller must s_wait_tensorcnt + barrier.
// ---------------------------------------------------------------------------
__device__ __forceinline__ void tdm_load_to_lds_1d(const _Float16* gsrc,
                                                   unsigned lds_off,
                                                   unsigned nHalf) {
    unsigned long long ga = (unsigned long long)(size_t)gsrc;
    u32x4 g0;
    g0[0] = 1u;                                    // count=1, user descriptor
    g0[1] = lds_off;                               // LDS byte address
    g0[2] = (unsigned)ga;                          // global_addr[31:0]
    g0[3] = ((unsigned)(ga >> 32) & 0x01ffffffu)   // global_addr[56:32]
          | (2u << 30);                            // type = 2 ("image")
    i32x8 g1;
    g1[0] = 1 << 16;                               // data_size = 1 -> 2 bytes
    g1[1] = (int)((nHalf & 0xffffu) << 16);        // tensor_dim0[15:0]
    g1[2] = (int)((nHalf >> 16) | (1u << 16));     // tensor_dim0[31:16]; tensor_dim1=1
    g1[3] = (int)(nHalf << 16);                    // tensor_dim1 hi=0; tile_dim0=nHalf
    g1[4] = 0;                                     // tile_dim1=0, tile_dim2=0 (unused)
    g1[5] = (int)nHalf;                            // tensor_dim0_stride lo
    g1[6] = 0;                                     // stride hi / dim1_stride lo
    g1[7] = 0;
    i32x4 gz = {0, 0, 0, 0};
#if defined(__clang_major__) && (__clang_major__ >= 23)
    i32x8 gz8 = {0, 0, 0, 0, 0, 0, 0, 0};
    __builtin_amdgcn_tensor_load_to_lds(g0, g1, gz, gz, gz8, 0);
#else
    __builtin_amdgcn_tensor_load_to_lds(g0, g1, gz, gz, 0);
#endif
}

// ---------------------------------------------------------------------------
// Pack a row-major f32 weight [N x Kreal] into f16 B-fragment layout,
// zero-padding K to Kpad. tile index = nT*(Kpad/32)+kT, 512 halfs per tile.
// ---------------------------------------------------------------------------
__global__ void pack_w(const float* __restrict__ W, _Float16* __restrict__ out,
                       int Kreal, int Kpad, int totalHalf) {
    int tid = blockIdx.x * blockDim.x + threadIdx.x;
    if (tid >= totalHalf) return;
    int e    = tid & 15;
    int lane = (tid >> 4) & 31;
    int tile = tid >> 9;
    int KTl  = Kpad >> 5;
    int nT = tile / KTl, kT = tile - nT * KTl;
    int n = nT * 16 + (lane & 15);
    int k = kT * 32 + frag_k(e, lane >> 4);
    float v = (k < Kreal) ? W[(size_t)n * Kreal + k] : 0.f;
    out[tid] = (_Float16)v;
}

// ---------------------------------------------------------------------------
// Embedding gather, written directly in packed A-fragment layout [MT x 10 tiles]
// (K = 320, last 20 cols zero). m = t*B + b matches [T,B] flattening.
// ---------------------------------------------------------------------------
__global__ void embed_pack(const int* __restrict__ text, const float* __restrict__ emb,
                           _Float16* __restrict__ x0) {
    int tid  = blockIdx.x * blockDim.x + threadIdx.x;     // exactly MT*10*512 threads
    int e    = tid & 15;
    int lane = (tid >> 4) & 31;
    int tile = tid >> 9;
    int mT = tile / 10, kT = tile - mT * 10;
    int m = mT * 16 + (lane & 15);
    int k = kT * 32 + frag_k(e, lane >> 4);
    float v = 0.f;
    if (k < EMB_D) {
        int tok = text[m];
        v = emb[(size_t)tok * EMB_D + k];
    }
    x0[tid] = (_Float16)v;
}

// ---------------------------------------------------------------------------
// xp = A * W^T + (b_ih + b_hh).  A: packed-A f16 [MT x KTl tiles],
// W: packed-B f16 [NT_H x KTl tiles].  Output packed-C f16 [MT x NT_H tiles].
// 8 waves/block share one mT strip: wave0 DMAs it to LDS via the TDM, then
// each wave K-loops v_wmma_f32_16x16x32_f16 with A from LDS, B from global/L2.
// ---------------------------------------------------------------------------
__global__ void __launch_bounds__(256) gemm_xp(
    const _Float16* __restrict__ A, const _Float16* __restrict__ Wp,
    const float* __restrict__ bias_ih, const float* __restrict__ bias_hh,
    _Float16* __restrict__ Cout, int KTl) {
    __shared__ alignas(32) _Float16 sA[32 * 512];   // up to 32 K-tiles = 32 KB

    const int wv   = threadIdx.x >> 5;
    const int lane = threadIdx.x & 31;
    const int mT = blockIdx.x >> 2;                 // 4 blocks span the 32 nT
    const int nT = (blockIdx.x & 3) * 8 + wv;
    if (mT >= MT) return;

    if (wv == 0) {
        tdm_load_to_lds_1d(A + (size_t)mT * KTl * 512,
                           (unsigned)(size_t)(void*)sA,
                           (unsigned)(KTl * 512));
        __builtin_amdgcn_s_wait_tensorcnt(0);
    }
    __syncthreads();

    const v16h* Bf = (const v16h*)(Wp + (size_t)nT * KTl * 512 + lane * 16);
    v8f acc = {};
    for (int kT = 0; kT < KTl; ++kT) {
        v16h a = *(const v16h*)(sA + kT * 512 + lane * 16);
        v16h b = Bf[kT * 32];   // 512 halfs per tile = 32 v16h
        acc = __builtin_amdgcn_wmma_f32_16x16x32_f16(false, a, false, b,
                                                     (short)0, acc, false, false);
    }
    int n = nT * 16 + (lane & 15);
    float bsum = bias_ih[n] + bias_hh[n];
    v8h outv;
#pragma unroll
    for (int r = 0; r < 8; ++r) outv[r] = (_Float16)(acc[r] + bsum);
    *(v8h*)(Cout + (size_t)(mT * NT_H + nT) * 256 + lane * 8) = outv;
}

// ---------------------------------------------------------------------------
// Recurrent scan: grid = 2 (dir), block = 1024 (32 waves). Persistent over t.
// h [64 x 512] lives in LDS in packed-A f16 layout (64 KB). Wave w owns output
// columns n in [16w,16w+16): per step it loads xp (packed-C), runs K=512 of
// h @ W_hh^T via 16 WMMA k-steps x 4 M-tiles, applies tanh + length masking,
// then scatters the new h back into the LDS A-layout (and optionally writes
// y into x1's packed-A layout, K offset dir*512 for the bidir concat).
// ---------------------------------------------------------------------------
__global__ void __launch_bounds__(1024) rnn_scan(
    const _Float16* __restrict__ xp,     // [2][MT*NT_H*256] packed-C f16
    const _Float16* __restrict__ WhhP,   // [2][NT_H*16*512] packed-B f16
    const int* __restrict__ lens,
    _Float16* __restrict__ x1out,        // packed-A, K=1024 (or unused)
    float* __restrict__ hfinal,          // [2][64][512] f32 (or null)
    int write_y)
{
    const int dir  = blockIdx.x;
    const int tid  = threadIdx.x;
    const int lane = tid & 31;
    const int w    = tid >> 5;           // nT owned by this wave

    __shared__ alignas(32) _Float16 hS[BATCH * HID];
    __shared__ int lensS[BATCH];
    for (int i = tid; i < BATCH * HID; i += 1024) hS[i] = (_Float16)0.f;
    if (tid < BATCH) lensS[tid] = lens[tid];
    __syncthreads();

    const _Float16* xpD = xp   + (size_t)dir * ((size_t)MT * NT_H * 256);
    const _Float16* WD  = WhhP + (size_t)dir * ((size_t)NT_H * 16 * 512);

    // C-layout coords for this lane; A-layout scatter coords for column k_h.
    const int hi_c = lane >> 4;
    const int col  = lane & 15;
    const int k_h  = w * 16 + col;       // hidden index this lane produces
    const int kT_h = k_h >> 5;
    int hi_a, e_a;
    frag_k_inv(k_h & 31, hi_a, e_a);

    for (int s = 0; s < T_LEN; ++s) {
        const int tt = dir ? (T_LEN - 1 - s) : s;

        v8f acc[4];
#pragma unroll
        for (int i = 0; i < 4; ++i) {
            v8h c = *(const v8h*)(xpD + (size_t)((tt*4 + i) * NT_H + w) * 256 + lane * 8);
#pragma unroll
            for (int r = 0; r < 8; ++r) acc[i][r] = (float)c[r];
        }

        for (int kT = 0; kT < 16; ++kT) {
            v16h b = *(const v16h*)(WD + (size_t)(w * 16 + kT) * 512 + lane * 16);
#pragma unroll
            for (int i = 0; i < 4; ++i) {
                v16h a = *(const v16h*)(hS + (i * 16 + kT) * 512 + lane * 16);
                acc[i] = __builtin_amdgcn_wmma_f32_16x16x32_f16(false, a, false, b,
                                                                (short)0, acc[i],
                                                                false, false);
            }
        }

        _Float16 nh[4][8], yv[4][8];
#pragma unroll
        for (int i = 0; i < 4; ++i) {
#pragma unroll
            for (int r = 0; r < 8; ++r) {
                int row  = r + 8 * hi_c;
                int bidx = 16 * i + row;
                bool valid = tt < lensS[bidx];
                float hn = tanhf(acc[i][r]);
                int lane_a = row + 16 * hi_a;
                int addr = (i * 16 + kT_h) * 512 + lane_a * 16 + e_a;
                _Float16 old = hS[addr];                 // pre-barrier read: safe
                nh[i][r] = valid ? (_Float16)hn : old;   // carry freezes
                yv[i][r] = valid ? (_Float16)hn : (_Float16)0.f;
            }
        }
        __syncthreads();   // everyone done reading h
#pragma unroll
        for (int i = 0; i < 4; ++i) {
#pragma unroll
            for (int r = 0; r < 8; ++r) {
                int row    = r + 8 * hi_c;
                int lane_a = row + 16 * hi_a;
                int addr = (i * 16 + kT_h) * 512 + lane_a * 16 + e_a;
                hS[addr] = nh[i][r];
                if (write_y) {
                    int mTg = tt * 4 + i;
                    int kTg = dir * 16 + kT_h;           // bidir concat along K
                    x1out[(size_t)(mTg * 32 + kTg) * 512 + lane_a * 16 + e_a] = yv[i][r];
                }
            }
        }
        __syncthreads();   // writes visible before next step's reads
    }

    if (hfinal) {
        for (int idx = tid; idx < BATCH * HID; idx += 1024) {
            int b = idx >> 9, k = idx & 511;
            int mTl = b >> 4, row = b & 15;
            int kT = k >> 5;
            int hi, e;
            frag_k_inv(k & 31, hi, e);
            hfinal[(size_t)dir * BATCH * HID + idx] =
                (float)hS[(mTl * 16 + kT) * 512 + (row + 16 * hi) * 16 + e];
        }
    }
}

// ---------------------------------------------------------------------------
// out[b] = [hf(b) | hb(b)] . fc_w + fc_b   (B=64, 2H=1024 -> tiny)
// ---------------------------------------------------------------------------
__global__ void fc_kernel(const float* __restrict__ hfin, const float* __restrict__ fc_w,
                          const float* __restrict__ fc_b, float* __restrict__ out) {
    int b = blockIdx.x;
    __shared__ float red[256];
    float s = 0.f;
    for (int k = threadIdx.x; k < HID; k += 256)
        s += hfin[b * HID + k] * fc_w[k]
           + hfin[BATCH * HID + b * HID + k] * fc_w[HID + k];
    red[threadIdx.x] = s;
    __syncthreads();
    for (int off = 128; off; off >>= 1) {
        if (threadIdx.x < off) red[threadIdx.x] += red[threadIdx.x + off];
        __syncthreads();
    }
    if (threadIdx.x == 0) out[b] = red[0] + fc_b[0];
}

// ---------------------------------------------------------------------------
extern "C" void kernel_launch(void* const* d_in, const int* in_sizes, int n_in,
                              void* d_out, int out_size, void* d_ws, size_t ws_size,
                              hipStream_t stream) {
    const int*   text    = (const int*)d_in[0];
    const int*   lens    = (const int*)d_in[1];
    const float* emb     = (const float*)d_in[2];
    const float* W_ih_l0 = (const float*)d_in[3];
    const float* W_hh_l0 = (const float*)d_in[4];
    const float* b_l0    = (const float*)d_in[5];
    const float* W_ih_l1 = (const float*)d_in[6];
    const float* W_hh_l1 = (const float*)d_in[7];
    const float* b_l1    = (const float*)d_in[8];
    const float* fc_w    = (const float*)d_in[9];
    const float* fc_b    = (const float*)d_in[10];
    float* out = (float*)d_out;

    char* ws = (char*)d_ws;
    size_t off = 0;
    auto take = [&](size_t bytes) {
        char* p = ws + off;
        off += (bytes + 255) & ~(size_t)255;
        return p;
    };
    _Float16* x0   = (_Float16*)take((size_t)M_TOT * EMB_P * 2);        // 21 MB
    _Float16* x1   = (_Float16*)take((size_t)M_TOT * 1024 * 2);         // 67 MB
    _Float16* xpb  = (_Float16*)take((size_t)2 * MT * NT_H * 256 * 2);  // 67 MB (reused l0->l1)
    _Float16* wih0 = (_Float16*)take((size_t)2 * 32 * 10 * 512 * 2);
    _Float16* whh0 = (_Float16*)take((size_t)2 * 32 * 16 * 512 * 2);
    _Float16* wih1 = (_Float16*)take((size_t)2 * 32 * 32 * 512 * 2);
    _Float16* whh1 = (_Float16*)take((size_t)2 * 32 * 16 * 512 * 2);
    float*    hfin = (float*)take((size_t)2 * BATCH * HID * 4);

    // Pack weights (f32 -> f16 B-fragment layout)
    for (int d = 0; d < 2; ++d) {
        pack_w<<<640,  256, 0, stream>>>(W_ih_l0 + (size_t)d * HID * EMB_D,
                                         wih0 + (size_t)d * 32 * 10 * 512,
                                         EMB_D, EMB_P, 32 * 10 * 512);
        pack_w<<<1024, 256, 0, stream>>>(W_hh_l0 + (size_t)d * HID * HID,
                                         whh0 + (size_t)d * 32 * 16 * 512,
                                         HID, HID, 32 * 16 * 512);
        pack_w<<<2048, 256, 0, stream>>>(W_ih_l1 + (size_t)d * HID * 1024,
                                         wih1 + (size_t)d * 32 * 32 * 512,
                                         1024, 1024, 32 * 32 * 512);
        pack_w<<<1024, 256, 0, stream>>>(W_hh_l1 + (size_t)d * HID * HID,
                                         whh1 + (size_t)d * 32 * 16 * 512,
                                         HID, HID, 32 * 16 * 512);
    }

    // Embedding gather -> packed-A f16 (MT*10*512 = 10,485,760 halfs)
    embed_pack<<<40960, 256, 0, stream>>>(text, emb, x0);

    // Layer 0 input projections
    for (int d = 0; d < 2; ++d)
        gemm_xp<<<8192, 256, 0, stream>>>(x0, wih0 + (size_t)d * 32 * 10 * 512,
                                          b_l0 + (size_t)d * 2 * HID,
                                          b_l0 + (size_t)d * 2 * HID + HID,
                                          xpb + (size_t)d * MT * NT_H * 256, 10);
    // Layer 0 scan (writes y into x1 packed-A layout)
    rnn_scan<<<2, 1024, 0, stream>>>(xpb, whh0, lens, x1, nullptr, 1);

    // Layer 1 input projections (xp buffer reused)
    for (int d = 0; d < 2; ++d)
        gemm_xp<<<8192, 256, 0, stream>>>(x1, wih1 + (size_t)d * 32 * 32 * 512,
                                          b_l1 + (size_t)d * 2 * HID,
                                          b_l1 + (size_t)d * 2 * HID + HID,
                                          xpb + (size_t)d * MT * NT_H * 256, 32);
    // Layer 1 scan (final hidden states only)
    rnn_scan<<<2, 1024, 0, stream>>>(xpb, whh1, lens, nullptr, hfin, 0);

    // Final FC
    fc_kernel<<<BATCH, 256, 0, stream>>>(hfin, fc_w, fc_b, out);
}
